// EnhancedCAREGNN_53206054863627
// MI455X (gfx1250) — compile-verified, matrix operand order checked by reading, not run on hardware
//
#include <hip/hip_runtime.h>
#include <hip/hip_bf16.h>

#define NN 100000   // nodes
#define DD 128      // input dim
#define DOUT 128    // output dim
#define VV 3        // views
#define EE 500000   // edges per view
#define CC 2        // classes
#define HH 64       // attention hidden (H == VH == 64)

typedef __attribute__((ext_vector_type(16))) _Float16 v16h;
typedef __attribute__((ext_vector_type(8)))  float    v8f;
typedef __attribute__((ext_vector_type(4)))  float    f32x4;
typedef __attribute__((ext_vector_type(4)))  _Float16 f16x4;

// ---------------------------------------------------------------------------
// WMMA fragment loader. LDS tiles are stored as [16-rows][32 k] halves with
// padded stride. 16-bit A-matrix 16x32 layout (ISA 7.12.2):
//   lanes 0-15 : row M=lane,    K pairs {0,1..6,7} then {16..23}
//   lanes 16-31: row M=lane-16, K pairs {8..15} and {24..31}
// B tile is staged K-packed per column so the same loader works for B.
// Lowered by the compiler to two ds_load_b128 per fragment.
// ---------------------------------------------------------------------------
__device__ inline v16h ldsFrag(const _Float16* base, int stride, int lane) {
  union { v16h v; unsigned u[8]; } f;
  const _Float16* p = base + (lane & 15) * stride;
  const int kh = (lane >> 4) << 3;  // 0 or 8
#pragma unroll
  for (int i = 0; i < 8; ++i) {
    const int k = (i < 4) ? (2 * i + kh) : (16 + 2 * (i - 4) + kh);
    f.u[i] = *(const unsigned*)(p + k);   // two consecutive halves, 4B aligned
  }
  return f.v;
}

// ---------------------------------------------------------------------------
// Tiled WMMA GEMM:  C[M,Nc] = op( A[M,K] (*colScale) @ B[K,Nc] + bias
//                                 + rowScale[m]*bias2[col] )
// Compile-time variants kill all nullable-pointer branching in the hot loop.
// A is split as A1 (k<128) / A2 (k>=128) for the concat GEMM (K=256).
// Out-of-range A rows are CLAMPED (not zero-padded): in D = A@B + C each
// output row depends only on its own A row, and rows >= M are never stored,
// so the clamp keeps the staging loop completely branch-free.
// Block tile: 128 rows x 64 cols, BK=32. 256 threads = 8 waves; each wave owns
// 16 rows x 64 cols = 4 WMMA accumulators.
// ---------------------------------------------------------------------------
#define BM 128
#define BK 32
#define LDT 40  // padded LDS stride in halves (80B rows: 8B-aligned, bank-skewed)

template <bool HAS_A2, bool HAS_CS, bool HAS_RS, bool RELU>
__global__ __launch_bounds__(256) void gemm_wmma_kernel(
    const float* __restrict__ A1, const float* __restrict__ A2,
    const float* __restrict__ colScale,
    const float* __restrict__ B,
    const float* __restrict__ bias,
    const float* __restrict__ bias2, const float* __restrict__ rowScale,
    float* __restrict__ Cout,
    int M, int K, int Nc) {
  __shared__ __align__(16) _Float16 As[BM * LDT];
  __shared__ __align__(16) _Float16 Bt[64 * LDT];

  const int tid  = threadIdx.x;
  const int lane = tid & 31;
  const int wave = tid >> 5;
  const int rBase = blockIdx.x * BM;
  const int nBase = blockIdx.y * 64;

  v8f acc[4] = {};

  for (int kBase = 0; kBase < K; kBase += BK) {
    // ---- Stage A tile (128 rows x 32 k) as float4 loads -> f16x4 LDS stores.
    // 1024 float4 transfers / 256 threads = 4 each; branch-free (row clamp).
#pragma unroll
    for (int i = 0; i < 4; ++i) {
      const int e  = tid + i * 256;    // 0..1023
      const int r  = e >> 3;           // 0..127
      const int k4 = (e & 7) * 4;      // 0,4,...,28
      int gr = rBase + r;
      gr = gr < M ? gr : (M - 1);      // clamp: rows >= M never stored
      const int gk = kBase + k4;
      const float* src = (HAS_A2 && gk >= 128)
                             ? (A2 + (long)gr * 128 + (gk - 128))
                             : (A1 + (long)gr * 128 + gk);
      f32x4 v = *(const f32x4*)src;    // 16B aligned (gk multiple of 4)
      if (HAS_CS) {
        v.x *= colScale[gk + 0];
        v.y *= colScale[gk + 1];
        v.z *= colScale[gk + 2];
        v.w *= colScale[gk + 3];
      }
      f16x4 h;
      h.x = (_Float16)v.x; h.y = (_Float16)v.y;
      h.z = (_Float16)v.z; h.w = (_Float16)v.w;
      *(f16x4*)(As + r * LDT + k4) = h;   // 8B-aligned ds_store_b64
    }
    // ---- Stage B tile (32 k x 64 n) as float4 loads, K-packed transposed
    // store: 512 float4 / 256 threads = 2 each.
#pragma unroll
    for (int i = 0; i < 2; ++i) {
      const int e  = tid + i * 256;    // 0..511
      const int k  = e >> 4;           // 0..31
      const int n4 = (e & 15) * 4;     // 0..60
      const f32x4 v = *(const f32x4*)(B + (long)(kBase + k) * Nc + nBase + n4);
      Bt[(n4 + 0) * LDT + k] = (_Float16)v.x;
      Bt[(n4 + 1) * LDT + k] = (_Float16)v.y;
      Bt[(n4 + 2) * LDT + k] = (_Float16)v.z;
      Bt[(n4 + 3) * LDT + k] = (_Float16)v.w;
    }
    // ---- Prefetch next A tile (global_prefetch_b8): one byte per thread
    // covers 128 rows x 2 cachelines while the WMMAs below execute.
    const int gk2 = kBase + BK;
    if (gk2 < K) {
      int pr = rBase + (tid >> 1);
      pr = pr < (M - 1) ? pr : (M - 1);
      const int pk = gk2 + (tid & 1) * 16;
      const float* pf = (HAS_A2 && gk2 >= 128)
                            ? (A2 + (long)pr * 128 + (pk - 128))
                            : (A1 + (long)pr * 128 + pk);
      __builtin_prefetch(pf, 0, 1);
    }
    __syncthreads();

    const v16h va = ldsFrag(As + (wave * 16) * LDT, LDT, lane);
#pragma unroll
    for (int nt = 0; nt < 4; ++nt) {
      const v16h vb = ldsFrag(Bt + (nt * 16) * LDT, LDT, lane);
      acc[nt] = __builtin_amdgcn_wmma_f32_16x16x32_f16(
          false, va, false, vb, (short)0, acc[nt], false, false);
    }
    __syncthreads();
  }

  // Epilogue. C/D layout: VGPR r -> (M = r + 8*laneHi, N = lane&15).
  const int rowOff = rBase + wave * 16 + ((lane >> 4) << 3);
#pragma unroll
  for (int nt = 0; nt < 4; ++nt) {
    const int col = nBase + nt * 16 + (lane & 15);
    const float badd = bias ? bias[col] : 0.f;
    const float b2 = HAS_RS ? bias2[col] : 0.f;
#pragma unroll
    for (int r = 0; r < 8; ++r) {
      const int row = rowOff + r;
      if (row < M) {
        float v = acc[nt][r] + badd;
        if (HAS_RS) v += rowScale[row] * b2;
        if (RELU) v = fmaxf(v, 0.f);
        Cout[(long)row * Nc + col] = v;
      }
    }
  }
}

// ---------------------------------------------------------------------------
// Edge stage: aggRaw[v][dst] += w * features[src]; wsum[v][dst] += w.
// One wave (32 lanes) per edge; each lane handles 4 of the 128 features.
// Both the 51 MB feature table and the 154 MB accumulator are L2-resident.
// ---------------------------------------------------------------------------
__global__ __launch_bounds__(256) void scatter_kernel(
    const float* __restrict__ feat, const int* __restrict__ esrc,
    const int* __restrict__ edst, const float* __restrict__ ew,
    float* __restrict__ aggRaw, float* __restrict__ wsum) {
  const long gid = (long)blockIdx.x * blockDim.x + threadIdx.x;
  const long wid = gid >> 5;
  const int lane = (int)(gid & 31);
  if (wid >= (long)VV * EE) return;
  const int v = (int)(wid / EE);
  const long e = wid % EE;
  const int s = esrc[(long)v * EE + e];
  const int d = edst[(long)v * EE + e];
  const float w = ew[(long)v * EE + e];
  const float* f = feat + (long)s * DD;
  float* o = aggRaw + ((long)v * NN + d) * DD;
#pragma unroll
  for (int i = 0; i < 4; ++i) {
    const int c = lane + i * 32;
    atomicAdd(o + c, w * f[c]);
  }
  if (lane == 0) atomicAdd(wsum + (long)v * NN + d, w);
}

// view_out = sigmoid(gateLin) * agg   (in-place into gateLin buffer)
__global__ void gate_apply_kernel(const float* __restrict__ agg,
                                  float* __restrict__ gateLin) {
  const long i = (long)blockIdx.x * blockDim.x + threadIdx.x;
  if (i >= (long)VV * NN * DOUT) return;
  const float x = gateLin[i];
  gateLin[i] = agg[i] / (1.f + __expf(-x));
}

// Label attention: class_probs + node_att (C=2, scalar math)
__global__ void label_att_kernel(
    const float* __restrict__ feat, const float* __restrict__ clsW,
    const float* __restrict__ clsb, const float* __restrict__ hBuf,
    const float* __restrict__ attW2, const float* __restrict__ attb2,
    const float* __restrict__ att_bias, float* __restrict__ node_att,
    float* __restrict__ probs_out) {
  const int n = blockIdx.x * blockDim.x + threadIdx.x;
  if (n >= NN) return;
  const float* f = feat + (long)n * DD;
  float l0 = clsb[0], l1 = clsb[1];
  for (int d = 0; d < DD; ++d) {
    const float fv = f[d];
    l0 += fv * clsW[d * 2 + 0];
    l1 += fv * clsW[d * 2 + 1];
  }
  const float m = fmaxf(l0, l1);
  const float e0 = __expf(l0 - m), e1 = __expf(l1 - m);
  const float inv = 1.f / (e0 + e1);
  const float p[2] = {e0 * inv, e1 * inv};
  float att = att_bias[0];
#pragma unroll
  for (int c = 0; c < CC; ++c) {
    const float* h = hBuf + ((long)c * NN + n) * HH;
    float s = attb2[c];
    for (int k = 0; k < HH; ++k) s += h[k] * attW2[c * HH + k];
    att += p[c] / (1.f + __expf(-s));  // sigmoid(s) * prob
  }
  node_att[n] = att;
  probs_out[(long)n * 2 + 0] = p[0];
  probs_out[(long)n * 2 + 1] = p[1];
}

// View-level softmax + combine + node-attention weighting. One block per node.
__global__ __launch_bounds__(128) void view_combine_kernel(
    const float* __restrict__ viewOut, const float* __restrict__ vh,
    const float* __restrict__ vattW2, const float* __restrict__ vattb2,
    const float* __restrict__ node_att, float* __restrict__ weighted) {
  const int n = blockIdx.x;
  const int o = threadIdx.x;
  __shared__ float sw[VV];
  if (o < VV) {
    const float* h = vh + ((long)o * NN + n) * HH;
    float s = vattb2[0];
    for (int k = 0; k < HH; ++k) s += h[k] * vattW2[k];
    sw[o] = s;
  }
  __syncthreads();
  const float m = fmaxf(sw[0], fmaxf(sw[1], sw[2]));
  const float e0 = __expf(sw[0] - m), e1 = __expf(sw[1] - m), e2 = __expf(sw[2] - m);
  const float inv = 1.f / (e0 + e1 + e2);
  const float comb = (viewOut[((long)0 * NN + n) * DOUT + o] * e0 +
                      viewOut[((long)1 * NN + n) * DOUT + o] * e1 +
                      viewOut[((long)2 * NN + n) * DOUT + o] * e2) * inv;
  weighted[(long)n * DOUT + o] = comb * node_att[n];
}

// out = LayerNorm(fused + transformed) * g + b.  One block (128 thr) per node.
__global__ __launch_bounds__(128) void final_ln_kernel(
    const float* __restrict__ fused, const float* __restrict__ transformed,
    const float* __restrict__ g, const float* __restrict__ b,
    float* __restrict__ out) {
  const int n = blockIdx.x, o = threadIdx.x;
  const float v = fused[(long)n * DOUT + o] + transformed[(long)n * DOUT + o];
  __shared__ float s1[128], s2[128];
  s1[o] = v;
  s2[o] = v * v;
  __syncthreads();
  for (int st = 64; st > 0; st >>= 1) {
    if (o < st) { s1[o] += s1[o + st]; s2[o] += s2[o + st]; }
    __syncthreads();
  }
  const float mu  = s1[0] * (1.f / 128.f);
  const float var = s2[0] * (1.f / 128.f) - mu * mu;
  out[(long)n * DOUT + o] = (v - mu) * rsqrtf(var + 1e-5f) * g[o] + b[o];
}

// ---------------------------------------------------------------------------
extern "C" void kernel_launch(void* const* d_in, const int* in_sizes, int n_in,
                              void* d_out, int out_size, void* d_ws, size_t ws_size,
                              hipStream_t stream) {
  (void)in_sizes; (void)n_in; (void)out_size; (void)ws_size;

  const float* feat      = (const float*)d_in[0];
  const int*   esrc      = (const int*)d_in[1];
  const int*   edst      = (const int*)d_in[2];
  const float* ew        = (const float*)d_in[3];
  const float* clsW      = (const float*)d_in[4];
  const float* clsb      = (const float*)d_in[5];
  const float* attW1     = (const float*)d_in[6];
  const float* attb1     = (const float*)d_in[7];
  const float* attW2     = (const float*)d_in[8];
  const float* attb2     = (const float*)d_in[9];
  const float* att_bias  = (const float*)d_in[10];
  const float* relW      = (const float*)d_in[11];
  const float* relb      = (const float*)d_in[12];
  const float* gateW     = (const float*)d_in[13];
  const float* gateb     = (const float*)d_in[14];
  const float* view_pref = (const float*)d_in[15];
  const float* vattW1    = (const float*)d_in[16];
  const float* vattb1    = (const float*)d_in[17];
  const float* vattW2    = (const float*)d_in[18];
  const float* vattb2    = (const float*)d_in[19];
  const float* selfW     = (const float*)d_in[20];
  const float* selfb     = (const float*)d_in[21];
  const float* featW     = (const float*)d_in[22];
  const float* featb     = (const float*)d_in[23];
  const float* fusW      = (const float*)d_in[24];
  const float* fusb      = (const float*)d_in[25];
  const float* ln_g      = (const float*)d_in[26];
  const float* ln_b      = (const float*)d_in[27];

  float* out       = (float*)d_out;                 // [N*128] then [N*2]
  float* probs_out = out + (size_t)NN * DOUT;

  // Workspace layout (floats)
  float* ws = (float*)d_ws;
  size_t off = 0;
  float* aggRaw  = ws + off; off += (size_t)VV * NN * DD;    // Σ w·feat[src]
  float* wsum    = ws + off; off += (size_t)VV * NN;         // Σ w
  float* agg     = ws + off; off += (size_t)VV * NN * DOUT;
  float* viewOut = ws + off; off += (size_t)VV * NN * DOUT;  // gateLin -> view_out
  float* vhB     = ws + off; off += (size_t)VV * NN * HH;
  float* hB      = ws + off; off += (size_t)CC * NN * HH;
  float* natt    = ws + off; off += (size_t)NN;
  float* selfO   = ws + off; off += (size_t)NN * DOUT;
  float* transf  = ws + off; off += (size_t)NN * DOUT;
  float* wght    = ws + off; off += (size_t)NN * DOUT;
  float* fusedB  = ws + off; off += (size_t)NN * DOUT;

  // Zero the atomic accumulators (aggRaw + wsum are contiguous).
  hipMemsetAsync(aggRaw, 0,
                 ((size_t)VV * NN * DD + (size_t)VV * NN) * sizeof(float), stream);

  // 1) Edge scatter: one wave per edge.
  {
    const long waves   = (long)VV * EE;
    const long threads = waves * 32;
    const unsigned blocks = (unsigned)((threads + 255) / 256);
    scatter_kernel<<<blocks, 256, 0, stream>>>(feat, esrc, edst, ew, aggRaw, wsum);
  }

  const dim3 g128((NN + BM - 1) / BM, DOUT / 64);
  const dim3 g64((NN + BM - 1) / BM, 1);

  // 2) agg[v] = aggRaw[v] @ relW[v] + wsum[v] ⊗ relb[v]
  for (int v = 0; v < VV; ++v)
    gemm_wmma_kernel<false, false, true, false><<<g128, 256, 0, stream>>>(
        aggRaw + (size_t)v * NN * DD, nullptr, nullptr,
        relW + (size_t)v * DD * DOUT, nullptr,
        relb + (size_t)v * DOUT, wsum + (size_t)v * NN,
        agg + (size_t)v * NN * DOUT, NN, DD, DOUT);

  // 3) gateLin[v] = agg[v] @ gateW[v] + gateb[v]
  for (int v = 0; v < VV; ++v)
    gemm_wmma_kernel<false, false, false, false><<<g128, 256, 0, stream>>>(
        agg + (size_t)v * NN * DOUT, nullptr, nullptr,
        gateW + (size_t)v * DOUT * DOUT, gateb + (size_t)v * DOUT,
        nullptr, nullptr,
        viewOut + (size_t)v * NN * DOUT, NN, DOUT, DOUT);

  // 4) view_out = sigmoid(gateLin) * agg
  {
    const long total = (long)VV * NN * DOUT;
    gate_apply_kernel<<<(unsigned)((total + 255) / 256), 256, 0, stream>>>(agg, viewOut);
  }

  // 5) vh[v] = relu((view_out[v] * view_pref[v]) @ vattW1 + vattb1)
  for (int v = 0; v < VV; ++v)
    gemm_wmma_kernel<false, true, false, true><<<g64, 256, 0, stream>>>(
        viewOut + (size_t)v * NN * DOUT, nullptr, view_pref + (size_t)v * DOUT,
        vattW1, vattb1, nullptr, nullptr,
        vhB + (size_t)v * NN * HH, NN, DOUT, HH);

  // 6) h[c] = relu(features @ attW1[c] + attb1[c])
  for (int c = 0; c < CC; ++c)
    gemm_wmma_kernel<false, false, false, true><<<g64, 256, 0, stream>>>(
        feat, nullptr, nullptr,
        attW1 + (size_t)c * DD * HH, attb1 + (size_t)c * HH,
        nullptr, nullptr,
        hB + (size_t)c * NN * HH, NN, DD, HH);

  // 7) self_out / transformed
  gemm_wmma_kernel<false, false, false, false><<<g128, 256, 0, stream>>>(
      feat, nullptr, nullptr, selfW, selfb, nullptr, nullptr, selfO, NN, DD, DOUT);
  gemm_wmma_kernel<false, false, false, false><<<g128, 256, 0, stream>>>(
      feat, nullptr, nullptr, featW, featb, nullptr, nullptr, transf, NN, DD, DOUT);

  // 8) label attention -> node_att + class_probs
  label_att_kernel<<<(NN + 255) / 256, 256, 0, stream>>>(
      feat, clsW, clsb, hB, attW2, attb2, att_bias, natt, probs_out);

  // 9) view softmax + combine + node-att weighting
  view_combine_kernel<<<NN, 128, 0, stream>>>(viewOut, vhB, vattW2, vattb2, natt, wght);

  // 10) fused = relu(concat(self_out, weighted) @ fusW + fusb)  (K=256 via A1/A2)
  gemm_wmma_kernel<true, false, false, true><<<g128, 256, 0, stream>>>(
      selfO, wght, nullptr, fusW, fusb, nullptr, nullptr, fusedB, NN, 2 * DOUT, DOUT);

  // 11) residual + LayerNorm -> out
  final_ln_kernel<<<NN, 128, 0, stream>>>(fusedB, transf, ln_g, ln_b, out);
}
